// SimpleMoELayer_21792664059952
// MI455X (gfx1250) — matrix operand order repaired and checked by reading, not compiled, not run on hardware
//
#include <hip/hip_runtime.h>

// ---------------- problem constants ----------------
#define T_TOK 8192   // B*S
#define DDIM  1024
#define HDIM  4096
#define NEXP  8
#define MT    64     // tokens per block tile (4 x 16 per wave)

typedef __bf16 v16bf __attribute__((ext_vector_type(16)));
typedef __bf16 v8bf  __attribute__((ext_vector_type(8)));
typedef __bf16 v2bf  __attribute__((ext_vector_type(2)));
typedef float  v8f   __attribute__((ext_vector_type(8)));
typedef int    v4i   __attribute__((ext_vector_type(4)));

struct alignas(16) U128 { unsigned int a, b, c, d; };
union Frag  { v16bf v; v2bf p[8]; unsigned short u[16]; U128 q[2]; };
union Half8 { v8bf  v; v2bf p[4]; unsigned short u[8];  U128 q; };

#if __has_builtin(__builtin_amdgcn_global_load_async_to_lds_b128) && \
    __has_builtin(__builtin_amdgcn_s_wait_asynccnt)
#define MOE_ASYNC_LDS 1
typedef __attribute__((address_space(1))) v4i ga_v4i;   // global (AS1)
typedef __attribute__((address_space(3))) v4i ls_v4i;   // LDS    (AS3)
#else
#define MOE_ASYNC_LDS 0
#endif

// 8x f32 -> 8x bf16 via packed converter (2 elems / instruction)
static __device__ __forceinline__ U128 cvt8(float4 a, float4 b) {
  Half8 h;
#if __has_builtin(__builtin_amdgcn_cvt_pk_bf16_f32)
  h.p[0] = __builtin_amdgcn_cvt_pk_bf16_f32(a.x, a.y);
  h.p[1] = __builtin_amdgcn_cvt_pk_bf16_f32(a.z, a.w);
  h.p[2] = __builtin_amdgcn_cvt_pk_bf16_f32(b.x, b.y);
  h.p[3] = __builtin_amdgcn_cvt_pk_bf16_f32(b.z, b.w);
#else
  h.v[0] = (__bf16)a.x; h.v[1] = (__bf16)a.y;
  h.v[2] = (__bf16)a.z; h.v[3] = (__bf16)a.w;
  h.v[4] = (__bf16)b.x; h.v[5] = (__bf16)b.y;
  h.v[6] = (__bf16)b.z; h.v[7] = (__bf16)b.w;
#endif
  return h.q;
}

static __device__ __forceinline__ unsigned short bf_bits(float f) {
  __bf16 h = (__bf16)f;
  return __builtin_bit_cast(unsigned short, h);
}

// build the 16-element bf16 B fragment from 16 contiguous f32 weights
static __device__ __forceinline__ Frag load_b_frag(const float* bp) {
  Frag b;
  float4 w0  = *(const float4*)(bp);
  float4 w1v = *(const float4*)(bp + 4);
  float4 w2v = *(const float4*)(bp + 8);
  float4 w3v = *(const float4*)(bp + 12);
  b.q[0] = cvt8(w0, w1v);
  b.q[1] = cvt8(w2v, w3v);
  return b;
}

// copy 32 bf16 (4 x b128) global -> LDS; async DMA path when available
static __device__ __forceinline__ void fill_seg(unsigned short* lds_dst,
                                                const unsigned short* gsrc) {
#if MOE_ASYNC_LDS
#pragma unroll
  for (int i = 0; i < 4; ++i)
    __builtin_amdgcn_global_load_async_to_lds_b128(
        (ga_v4i*)(gsrc + i * 8), (ls_v4i*)(lds_dst + i * 8), 0, 0);
#else
#pragma unroll
  for (int i = 0; i < 4; ++i)
    *(U128*)(lds_dst + i * 8) = *(const U128*)(gsrc + i * 8);
#endif
}

template <int PENDING>
static __device__ __forceinline__ void wait_fill() {
#if MOE_ASYNC_LDS
  __builtin_amdgcn_s_wait_asynccnt(PENDING);
#endif
}

// ---------------- kernel 0: zero expert counters ----------------
__global__ void moe_zero_counts(int* __restrict__ counts) {
  if (threadIdx.x < NEXP) counts[threadIdx.x] = 0;
}

// ---------------- kernel 1: x f32 -> bf16 ----------------
__global__ void __launch_bounds__(256)
moe_cvt_x(const float* __restrict__ x, unsigned short* __restrict__ xbf) {
  size_t i = ((size_t)blockIdx.x * 256 + threadIdx.x) * 8;
  float4 a = *(const float4*)(x + i);
  float4 b = *(const float4*)(x + i + 4);
  *(U128*)(xbf + i) = cvt8(a, b);
}

// ---------------- kernel 2: gate + top1 routing ----------------
__global__ void __launch_bounds__(256)
moe_gate(const float* __restrict__ x, const float* __restrict__ gw,
         const float* __restrict__ gb, int* __restrict__ counts,
         int* __restrict__ buckets) {
  const int wave = threadIdx.x >> 5, lane = threadIdx.x & 31;
  const int t = blockIdx.x * 8 + wave;
  if (t >= T_TOK) return;
  const float* xr = x + (size_t)t * DDIM;
  float acc[NEXP];
#pragma unroll
  for (int e = 0; e < NEXP; ++e) acc[e] = 0.f;
  for (int j = lane; j < DDIM; j += 32) {
    float xv = xr[j];
#pragma unroll
    for (int e = 0; e < NEXP; ++e) acc[e] += xv * gw[e * DDIM + j];
  }
#pragma unroll
  for (int off = 16; off > 0; off >>= 1) {
#pragma unroll
    for (int e = 0; e < NEXP; ++e) acc[e] += __shfl_xor(acc[e], off, 32);
  }
  if (lane == 0) {
    float best = acc[0] + gb[0];
    int be = 0;
#pragma unroll
    for (int e = 1; e < NEXP; ++e) {
      float v = acc[e] + gb[e];
      if (v > best) { best = v; be = e; }    // first-max tie-break
    }
    int pos = atomicAdd(&counts[be], 1);
    buckets[be * T_TOK + pos] = t;
  }
}

// ---------------- grouped GEMM: Out = act(A W^T + bias) ----------------
// A: bf16 [T_TOK x KDIM] gathered by token buckets. W: f32 [NEXP x NDIM x KDIM].
// grid.x = NEXP * (T_TOK/MT) row tiles, grid.y = NDIM/256 col tiles.
// 256 threads = 8 waves; wave w owns cols {n0+16w, n0+128+16w} and 4 M-subtiles:
// per k-step, 4 A frags x 2 B frags feed 8 WMMAs (each A frag has 2 uses).
// Double-buffered LDS A tile, async global->LDS fill overlapped with compute.
template <int KDIM, int NDIM, bool RELU_BF16_OUT>
__global__ void __launch_bounds__(256)
moe_gemm(const unsigned short* __restrict__ Abf, const float* __restrict__ W,
         const float* __restrict__ bias_vec, const int* __restrict__ counts,
         const int* __restrict__ buckets, void* __restrict__ outv) {
  const int e   = blockIdx.x >> 7;            // T_TOK/MT = 128 tiles/expert
  const int m0  = (blockIdx.x & 127) << 6;
  const int cnt = counts[e];
  if (m0 >= cnt) return;

  const int tid = threadIdx.x;
  const int wave = tid >> 5, lane = tid & 31;
  const int n0 = blockIdx.y * 256;

  __shared__ alignas(16) unsigned short sA[2][MT * 128];   // 2 x 16 KB
  __shared__ int sTok[MT];

  if (tid < MT) {
    int rr = (m0 + tid < cnt) ? tid : 0;      // duplicate row 0 on ragged tile
    sTok[tid] = buckets[e * T_TOK + m0 + rr];
  }
  __syncthreads();

  const int nw = n0 + wave * 16;
  const int nl = lane & 15;
  const int khA = (lane < 16) ? 0 : 16;       // A frag: K 0..15 / 16..31 contiguous
  const int khB = (lane < 16) ? 0 : 16;       // B frag: K 0..15 / 16..31 contiguous
  const float* wrow0 = W + ((size_t)e * NDIM + (nw + nl)) * KDIM;
  const float* wrow1 = wrow0 + (size_t)128 * KDIM;

  const float bias0 = bias_vec[(size_t)e * NDIM + nw + nl];
  const float bias1 = bias_vec[(size_t)e * NDIM + nw + 128 + nl];
  v8f acc[2][4];
#pragma unroll
  for (int mi = 0; mi < 4; ++mi)
#pragma unroll
    for (int i = 0; i < 8; ++i) { acc[0][mi][i] = bias0; acc[1][mi][i] = bias1; }

  const int frow = tid >> 2;                  // 0..63: LDS fill row
  const int fcol = (tid & 3) << 5;            // 0,32,64,96: 32 k per thread
  const int rowA = lane & 15;
  const unsigned short* arow = Abf + (size_t)sTok[frow] * KDIM + fcol;
  unsigned short* myfill0 = &sA[0][frow * 128 + fcol];
  unsigned short* myfill1 = &sA[1][frow * 128 + fcol];

  constexpr int NCHUNK = KDIM / 128;
  fill_seg(myfill0, arow);                    // prologue fill of buffer 0

  for (int c = 0; c < NCHUNK; ++c) {
    const int k0 = c * 128;
    if (c + 1 < NCHUNK) {
      fill_seg((c & 1) ? myfill0 : myfill1, arow + k0 + 128);
      wait_fill<4>();                         // previous fill done, next in flight
    } else {
      wait_fill<0>();
    }
    __syncthreads();                          // current buffer visible to all

    const unsigned short* buf = sA[c & 1];
#pragma unroll
    for (int kk = 0; kk < 128; kk += 32) {
      Frag b0 = load_b_frag(wrow0 + k0 + kk + khB);
      Frag b1 = load_b_frag(wrow1 + k0 + kk + khB);
      Frag a[4];
#pragma unroll
      for (int mi = 0; mi < 4; ++mi) {         // load all A frags first...
        const unsigned short* ap = &buf[(mi * 16 + rowA) * 128 + kk + khA];
        a[mi].q[0] = *(const U128*)(ap);
        a[mi].q[1] = *(const U128*)(ap + 8);
      }
#pragma unroll
      for (int mi = 0; mi < 4; ++mi) {         // ...then 8 WMMAs (2 uses per A frag)
        acc[0][mi] = __builtin_amdgcn_wmma_f32_16x16x32_bf16(
            false, a[mi].v, false, b0.v, (short)0, acc[0][mi], false, false);
        acc[1][mi] = __builtin_amdgcn_wmma_f32_16x16x32_bf16(
            false, a[mi].v, false, b1.v, (short)0, acc[1][mi], false, false);
      }
    }
    __syncthreads();                          // all reads done before overwrite
  }

  // epilogue
#pragma unroll
  for (int half = 0; half < 2; ++half) {
    const int ncol = nw + half * 128 + nl;
#pragma unroll
    for (int mi = 0; mi < 4; ++mi) {
#pragma unroll
      for (int v = 0; v < 8; ++v) {
        int M = mi * 16 + v + ((lane < 16) ? 0 : 8);
        if (m0 + M < cnt) {
          float val = acc[half][mi][v];
          if constexpr (RELU_BF16_OUT) {
            float r = val > 0.f ? val : 0.f;
            ((unsigned short*)outv)[(size_t)sTok[M] * NDIM + ncol] = bf_bits(r);
          } else {
            ((float*)outv)[(size_t)sTok[M] * NDIM + ncol] = val;
          }
        }
      }
    }
  }
}

// ---------------- launcher ----------------
extern "C" void kernel_launch(void* const* d_in, const int* in_sizes, int n_in,
                              void* d_out, int out_size, void* d_ws, size_t ws_size,
                              hipStream_t stream) {
  const float* x      = (const float*)d_in[0];
  const float* gate_w = (const float*)d_in[1];
  const float* gate_b = (const float*)d_in[2];
  const float* w1     = (const float*)d_in[3];
  const float* b1     = (const float*)d_in[4];
  const float* w2     = (const float*)d_in[5];
  const float* b2     = (const float*)d_in[6];
  float* out = (float*)d_out;

  // workspace: counts | buckets | x bf16 | H bf16   (~84.4 MB total)
  int* counts  = (int*)d_ws;
  int* buckets = (int*)((char*)d_ws + 256);
  unsigned short* Xbf  = (unsigned short*)((char*)d_ws + (size_t)(1 << 19));
  unsigned short* Hbuf = (unsigned short*)((char*)d_ws + (size_t)(1 << 19) +
                                           (size_t)T_TOK * DDIM * 2);

  moe_zero_counts<<<1, 32, 0, stream>>>(counts);
  moe_cvt_x<<<(T_TOK * DDIM) / (256 * 8), 256, 0, stream>>>(x, Xbf);
  moe_gate<<<T_TOK / 8, 256, 0, stream>>>(x, gate_w, gate_b, counts, buckets);
  dim3 g1(NEXP * (T_TOK / MT), HDIM / 256);
  moe_gemm<DDIM, HDIM, true><<<g1, 256, 0, stream>>>(
      Xbf, w1, b1, counts, buckets, (void*)Hbuf);
  dim3 g2(NEXP * (T_TOK / MT), DDIM / 256);
  moe_gemm<HDIM, DDIM, false><<<g2, 256, 0, stream>>>(
      Hbuf, w2, b2, counts, buckets, (void*)out);
  (void)in_sizes; (void)n_in; (void)out_size; (void)ws_size;
}
